// SwinTransformerV2Block_68693706932710
// MI455X (gfx1250) — compile-verified, matrix operand order checked by reading, not run on hardware
//
#include <hip/hip_runtime.h>
#include <hip/hip_bf16.h>
#include <math.h>

// ---------------------------------------------------------------------------
// Swin block on gfx1250: all GEMMs via v_wmma_f32_16x16x32_bf16 (wave32).
// B=32,H=W=56,C=256,WS=7,NH=8,HD=32,N=49(->pad 64),HID=1024, windows=2048.
// Round-2 changes: explicit register staging of reused fragments (no scratch
// spills), __launch_bounds__(256,1), chunked-K accumulation in FC2.
// ---------------------------------------------------------------------------

typedef __attribute__((ext_vector_type(16))) __bf16 bf16x16;
typedef __attribute__((ext_vector_type(8)))  float  floatx8;

#define C_     256
#define NPAD_  64
#define NTOK_  49
#define NH_    8
#define HID_   1024
#define NWIN_  2048
#define SCALE_ 0.17677669529663687f   // 32^-0.5
#define EPS_   1e-5f

static __device__ __forceinline__ floatx8 wmma_bf16(bf16x16 a, bf16x16 b, floatx8 c) {
  return __builtin_amdgcn_wmma_f32_16x16x32_bf16(false, a, false, b, (short)0, c, false, false);
}

static __device__ __forceinline__ floatx8 fzero8() {
  floatx8 z = {0.f,0.f,0.f,0.f,0.f,0.f,0.f,0.f};
  return z;
}

// Load a 16x32 bf16 A/B fragment from a row-major [rows x ld] buffer.
// p points at (row_base, k_base). ISA layout: lane l -> row (l&15);
// elements 0..7 = k 8*(l>>4)+0..7 ; elements 8..15 = k 16+8*(l>>4)+0..7.
static __device__ __forceinline__ bf16x16 frag_ld(const __bf16* p, int ld, int lane) {
  const __bf16* q = p + (lane & 15) * ld + ((lane >> 4) << 3);
  bf16x16 f;
  ((uint4*)&f)[0] = *(const uint4*)(q);
  ((uint4*)&f)[1] = *(const uint4*)(q + 16);
  return f;
}

static __device__ __forceinline__ float red_sum32(float v) {
  #pragma unroll
  for (int m = 1; m < 32; m <<= 1) v += __shfl_xor(v, m, 32);
  return v;
}
static __device__ __forceinline__ float red_sum16(float v) {
  #pragma unroll
  for (int m = 1; m < 16; m <<= 1) v += __shfl_xor(v, m, 32);
  return v;
}
static __device__ __forceinline__ float red_max16(float v) {
  #pragma unroll
  for (int m = 1; m < 16; m <<= 1) v = fmaxf(v, __shfl_xor(v, m, 32));
  return v;
}

// ---------------------------------------------------------------------------
// Kernel 0: convert all weight matrices fp32 -> bf16 once.
// ---------------------------------------------------------------------------
__global__ void k_convert_w(const float* __restrict__ qkv, const float* __restrict__ proj,
                            const float* __restrict__ fc1, const float* __restrict__ fc2,
                            __bf16* __restrict__ oqkv, __bf16* __restrict__ oproj,
                            __bf16* __restrict__ ofc1, __bf16* __restrict__ ofc2) {
  int i = blockIdx.x * blockDim.x + threadIdx.x;
  int stride = gridDim.x * blockDim.x;
  for (int k = i; k < 768 * 256;  k += stride) oqkv[k] = (__bf16)qkv[k];
  for (int k = i; k < 256 * 256;  k += stride) oproj[k] = (__bf16)proj[k];
  for (int k = i; k < 1024 * 256; k += stride) ofc1[k] = (__bf16)fc1[k];
  for (int k = i; k < 256 * 1024; k += stride) ofc2[k] = (__bf16)fc2[k];
}

// ---------------------------------------------------------------------------
// Kernel 1: LN1 + window partition -> xw bf16 [2048][64][256], pad rows = 0.
// ---------------------------------------------------------------------------
__global__ void __launch_bounds__(256, 1)
k_ln1_partition(const float* __restrict__ x,
                const float* __restrict__ g1, const float* __restrict__ b1,
                __bf16* __restrict__ xw) {
  const int w = blockIdx.x;
  const int lane = threadIdx.x & 31;
  const int wv = threadIdx.x >> 5;
  const int bi = w >> 6, wi = (w >> 3) & 7, wj = w & 7;
  for (int t = wv; t < NPAD_; t += 8) {
    __bf16* dst = xw + ((size_t)w * NPAD_ + t) * C_ + lane * 8;
    if (t < NTOK_) {
      int tok = bi * 3136 + (wi * 7 + t / 7) * 56 + (wj * 7 + t % 7);
      const float* xp = x + (size_t)tok * C_ + lane * 8;
      float4 va = *(const float4*)xp;
      float4 vb = *(const float4*)(xp + 4);
      float v[8] = {va.x, va.y, va.z, va.w, vb.x, vb.y, vb.z, vb.w};
      float s = 0.f;
      #pragma unroll
      for (int e = 0; e < 8; ++e) s += v[e];
      float mu = red_sum32(s) * (1.f / 256.f);
      float q = 0.f;
      #pragma unroll
      for (int e = 0; e < 8; ++e) { float d = v[e] - mu; q += d * d; }
      float var = red_sum32(q) * (1.f / 256.f);
      float rstd = rsqrtf(var + EPS_);
      __bf16 ob[8];
      #pragma unroll
      for (int e = 0; e < 8; ++e) {
        int c = lane * 8 + e;
        ob[e] = (__bf16)((v[e] - mu) * rstd * g1[c] + b1[c]);
      }
      *(uint4*)dst = *(uint4*)ob;
    } else {
      uint4 z = {0u, 0u, 0u, 0u};
      *(uint4*)dst = z;
    }
  }
}

// ---------------------------------------------------------------------------
// Kernel 2: fused attention. 1 block = 1 window (8 waves = 8 heads).
// Dynamic LDS: xw tile 32KB + per-head {Q 4K, K 4K, Vt 4K, P 8K} * 8 = 192KB.
// ---------------------------------------------------------------------------
__global__ void __launch_bounds__(256, 1)
k_attention(const __bf16* __restrict__ xw,
            const __bf16* __restrict__ wqkv,
            const float* __restrict__ qkv_b,
            const float* __restrict__ rpb,
            __bf16* __restrict__ ao) {
  extern __shared__ char smem[];
  __bf16* xs = (__bf16*)smem;                              // 64x256
  const int w = blockIdx.x;
  const int tid = threadIdx.x;
  const int lane = tid & 31;
  const int h = tid >> 5;                                  // wave id == head
  const int nl = lane & 15, hf = lane >> 4;

  { // cooperative stage of the window tile
    const uint4* src = (const uint4*)(xw + (size_t)w * NPAD_ * C_);
    uint4* dst = (uint4*)xs;
    for (int i = tid; i < (NPAD_ * C_ * 2) / 16; i += 256) dst[i] = src[i];
  }
  __syncthreads();

  __bf16* qs  = (__bf16*)(smem + 32768 + h * 20480);       // 64x32
  __bf16* ks  = qs + 2048;                                 // 64x32
  __bf16* vts = qs + 4096;                                 // 32x64 (V transposed)
  __bf16* ps  = qs + 6144;                                 // 64x64

  // --- QKV = xs @ Wqkv[head]^T + b ---
  // B-frags staged in registers per k-chunk (reused across 4 M-tiles);
  // 4 accumulators live; no scratch spills.
  #pragma unroll 1
  for (int s = 0; s < 3; ++s) {
    #pragma unroll 1
    for (int nt = 0; nt < 2; ++nt) {
      const __bf16* wrow = wqkv + (size_t)(s * 256 + h * 32 + nt * 16) * C_;
      floatx8 acc[4] = {fzero8(), fzero8(), fzero8(), fzero8()};
      #pragma unroll 1
      for (int kh = 0; kh < 2; ++kh) {
        bf16x16 bfr[4];
        #pragma unroll
        for (int kk = 0; kk < 4; ++kk)
          bfr[kk] = frag_ld(wrow + (kh * 4 + kk) * 32, C_, lane);
        #pragma unroll
        for (int mt = 0; mt < 4; ++mt) {
          #pragma unroll
          for (int kk = 0; kk < 4; ++kk) {
            bf16x16 a = frag_ld(xs + mt * 16 * C_ + (kh * 4 + kk) * 32, C_, lane);
            acc[mt] = wmma_bf16(a, bfr[kk], acc[mt]);
          }
        }
      }
      float bias = qkv_b[s * 256 + h * 32 + nt * 16 + nl];
      #pragma unroll
      for (int mt = 0; mt < 4; ++mt) {
        if (s == 0) {
          #pragma unroll
          for (int g = 0; g < 8; ++g)
            qs[(mt * 16 + g + 8 * hf) * 32 + nt * 16 + nl] = (__bf16)(acc[mt][g] + bias);
        } else if (s == 1) {
          #pragma unroll
          for (int g = 0; g < 8; ++g)
            ks[(mt * 16 + g + 8 * hf) * 32 + nt * 16 + nl] = (__bf16)(acc[mt][g] + bias);
        } else {
          #pragma unroll
          for (int g = 0; g < 8; ++g)   // transposed: vts[d][m]
            vts[(nt * 16 + nl) * 64 + mt * 16 + g + 8 * hf] = (__bf16)(acc[mt][g] + bias);
        }
      }
    }
  }
  // wave-private LDS; per-wave DS ordering suffices (no barrier needed).

  // --- S = scale*Q@K^T + relpos bias ; softmax ; P -> ps ---
  #pragma unroll 1
  for (int mt = 0; mt < 4; ++mt) {
    floatx8 sreg[4];
    bf16x16 aq = frag_ld(qs + mt * 16 * 32, 32, lane);
    #pragma unroll
    for (int nt = 0; nt < 4; ++nt) {
      bf16x16 bk = frag_ld(ks + nt * 16 * 32, 32, lane);
      sreg[nt] = wmma_bf16(aq, bk, fzero8());
    }
    #pragma unroll
    for (int nt = 0; nt < 4; ++nt) {
      int col = nt * 16 + nl;                       // key token
      #pragma unroll
      for (int g = 0; g < 8; ++g) {
        int row = mt * 16 + g + 8 * hf;             // query token
        float v = sreg[nt][g] * SCALE_;
        if (col < NTOK_) {
          if (row < NTOK_) {
            int ridx = ((row / 7) - (col / 7) + 6) * 13 + ((row % 7) - (col % 7) + 6);
            v += rpb[ridx * NH_ + h];
          }
        } else {
          v = -1e30f;                               // mask pad keys
        }
        sreg[nt][g] = v;
      }
    }
    #pragma unroll
    for (int g = 0; g < 8; ++g) {
      // row (mt*16+g+8*hf) lives on the 16 lanes sharing hf: reduce over nl x nt
      float m = fmaxf(fmaxf(sreg[0][g], sreg[1][g]), fmaxf(sreg[2][g], sreg[3][g]));
      m = red_max16(m);
      float e0 = __expf(sreg[0][g] - m);
      float e1 = __expf(sreg[1][g] - m);
      float e2 = __expf(sreg[2][g] - m);
      float e3 = __expf(sreg[3][g] - m);
      float sum = red_sum16(e0 + e1 + e2 + e3);
      float inv = 1.0f / sum;                       // sum >= 1 always
      int row = mt * 16 + g + 8 * hf;
      ps[row * 64 +  0 + nl] = (__bf16)(e0 * inv);
      ps[row * 64 + 16 + nl] = (__bf16)(e1 * inv);
      ps[row * 64 + 32 + nl] = (__bf16)(e2 * inv);
      ps[row * 64 + 48 + nl] = (__bf16)(e3 * inv);
    }
  }

  // --- O = P @ V  -> ao (window layout, bf16) ---
  #pragma unroll 1
  for (int mt = 0; mt < 4; ++mt) {
    #pragma unroll
    for (int nt = 0; nt < 2; ++nt) {
      floatx8 acc = fzero8();
      #pragma unroll
      for (int kk = 0; kk < 2; ++kk) {
        bf16x16 a = frag_ld(ps + mt * 16 * 64 + kk * 32, 64, lane);
        bf16x16 b = frag_ld(vts + nt * 16 * 64 + kk * 32, 64, lane);
        acc = wmma_bf16(a, b, acc);
      }
      __bf16* dst = ao + (size_t)w * NPAD_ * C_ + h * 32 + nt * 16 + nl;
      #pragma unroll
      for (int g = 0; g < 8; ++g)
        dst[(mt * 16 + g + 8 * hf) * C_] = (__bf16)acc[g];
    }
  }
}

// ---------------------------------------------------------------------------
// Kernel 3: proj GEMM + bias + window reverse + shortcut -> out (fp32).
// 1 block = 1 window. Wave wv owns M-tile (wv>>1) and N-tiles (wv&1)*8..+7.
// A-frags (shared across the wave's 8 tiles) staged once in registers.
// ---------------------------------------------------------------------------
__global__ void __launch_bounds__(256, 1)
k_proj(const __bf16* __restrict__ ao,
       const __bf16* __restrict__ wproj,
       const float* __restrict__ proj_b,
       const float* __restrict__ x,
       float* __restrict__ out) {
  __shared__ __bf16 as[NPAD_ * C_];
  const int w = blockIdx.x;
  const int tid = threadIdx.x, lane = tid & 31, wv = tid >> 5;
  const int nl = lane & 15, hf = lane >> 4;
  {
    const uint4* src = (const uint4*)(ao + (size_t)w * NPAD_ * C_);
    uint4* dst = (uint4*)as;
    for (int i = tid; i < (NPAD_ * C_ * 2) / 16; i += 256) dst[i] = src[i];
  }
  __syncthreads();
  const int bi = w >> 6, wi = (w >> 3) & 7, wj = w & 7;
  const int mt = wv >> 1;
  bf16x16 afr[8];
  #pragma unroll
  for (int kk = 0; kk < 8; ++kk)
    afr[kk] = frag_ld(as + mt * 16 * C_ + kk * 32, C_, lane);
  #pragma unroll 1
  for (int tt = 0; tt < 8; ++tt) {
    int nt = (wv & 1) * 8 + tt;
    floatx8 acc = fzero8();
    const __bf16* wrow = wproj + (size_t)(nt * 16) * C_;
    #pragma unroll
    for (int kk = 0; kk < 8; ++kk)
      acc = wmma_bf16(afr[kk], frag_ld(wrow + kk * 32, C_, lane), acc);
    int n = nt * 16 + nl;
    float pb = proj_b[n];
    #pragma unroll
    for (int g = 0; g < 8; ++g) {
      int row = mt * 16 + g + 8 * hf;
      if (row < NTOK_) {
        int tok = bi * 3136 + (wi * 7 + row / 7) * 56 + (wj * 7 + row % 7);
        out[(size_t)tok * C_ + n] = acc[g] + pb + x[(size_t)tok * C_ + n];
      }
    }
  }
}

// ---------------------------------------------------------------------------
// Kernel 4: fused MLP per 64-token block: LN2 -> FC1+GELU (LDS) -> FC2 +
// residual, in-place on out (each element read-then-written by one thread).
// Dynamic LDS: h 64x256 bf16 (32KB) + hid 64x1024 bf16 (128KB) = 160KB.
// ---------------------------------------------------------------------------
__global__ void __launch_bounds__(256, 1)
k_mlp(const float* __restrict__ g2, const float* __restrict__ b2,
      const __bf16* __restrict__ wfc1, const float* __restrict__ fc1_b,
      const __bf16* __restrict__ wfc2, const float* __restrict__ fc2_b,
      float* __restrict__ out) {
  extern __shared__ char smem[];
  __bf16* hs  = (__bf16*)smem;               // 64x256
  __bf16* hid = (__bf16*)(smem + 32768);     // 64x1024
  const size_t t0 = (size_t)blockIdx.x * 64;
  const int tid = threadIdx.x, lane = tid & 31, wv = tid >> 5;
  const int nl = lane & 15, hf = lane >> 4;

  // LN2 (rows wv*8 .. wv*8+7)
  #pragma unroll 1
  for (int q = 0; q < 8; ++q) {
    int m = wv * 8 + q;
    const float* xp = out + (t0 + m) * C_ + lane * 8;
    float4 va = *(const float4*)xp;
    float4 vb = *(const float4*)(xp + 4);
    float v[8] = {va.x, va.y, va.z, va.w, vb.x, vb.y, vb.z, vb.w};
    float s = 0.f;
    #pragma unroll
    for (int e = 0; e < 8; ++e) s += v[e];
    float mu = red_sum32(s) * (1.f / 256.f);
    float qq = 0.f;
    #pragma unroll
    for (int e = 0; e < 8; ++e) { float d = v[e] - mu; qq += d * d; }
    float rstd = rsqrtf(red_sum32(qq) * (1.f / 256.f) + EPS_);
    __bf16 ob[8];
    #pragma unroll
    for (int e = 0; e < 8; ++e) {
      int c = lane * 8 + e;
      ob[e] = (__bf16)((v[e] - mu) * rstd * g2[c] + b2[c]);
    }
    *(uint4*)(hs + m * C_ + lane * 8) = *(uint4*)ob;
  }
  __syncthreads();

  // FC1 + exact GELU. Wave wv owns M-tile (wv>>1), N-tiles (wv&1)*32..+31.
  // A-frags (K=256 -> 8 frags) staged once, reused across 32 N-tiles.
  {
    const int mt = wv >> 1;
    bf16x16 afr[8];
    #pragma unroll
    for (int kk = 0; kk < 8; ++kk)
      afr[kk] = frag_ld(hs + mt * 16 * C_ + kk * 32, C_, lane);
    #pragma unroll 1
    for (int tt = 0; tt < 32; ++tt) {
      int nt = (wv & 1) * 32 + tt;
      floatx8 acc = fzero8();
      const __bf16* wrow = wfc1 + (size_t)(nt * 16) * C_;
      #pragma unroll
      for (int kk = 0; kk < 8; ++kk)
        acc = wmma_bf16(afr[kk], frag_ld(wrow + kk * 32, C_, lane), acc);
      int n = nt * 16 + nl;
      float bias = fc1_b[n];
      #pragma unroll
      for (int g = 0; g < 8; ++g) {
        float u = acc[g] + bias;
        u = 0.5f * u * (1.0f + erff(u * 0.70710678118654752f));
        hid[(mt * 16 + g + 8 * hf) * HID_ + n] = (__bf16)u;
      }
    }
  }
  __syncthreads();

  // FC2 + bias + residual. Wave wv: M-tile (wv>>1), N-tiles (wv&1)*8..+7.
  // K=1024 chunked by 4 WMMA steps; 8 tile accumulators live for A reuse.
  {
    const int mt = wv >> 1;
    const int ntb = (wv & 1) * 8;
    floatx8 acc[8];
    #pragma unroll
    for (int tt = 0; tt < 8; ++tt) acc[tt] = fzero8();
    #pragma unroll 1
    for (int kh = 0; kh < 8; ++kh) {
      bf16x16 afr[4];
      #pragma unroll
      for (int kk = 0; kk < 4; ++kk)
        afr[kk] = frag_ld(hid + mt * 16 * HID_ + (kh * 4 + kk) * 32, HID_, lane);
      #pragma unroll
      for (int tt = 0; tt < 8; ++tt) {
        const __bf16* wrow = wfc2 + (size_t)((ntb + tt) * 16) * HID_;
        #pragma unroll
        for (int kk = 0; kk < 4; ++kk)
          acc[tt] = wmma_bf16(afr[kk], frag_ld(wrow + (kh * 4 + kk) * 32, HID_, lane), acc[tt]);
      }
    }
    #pragma unroll 1
    for (int tt = 0; tt < 8; ++tt) {
      int n = (ntb + tt) * 16 + nl;
      float bias = fc2_b[n];
      #pragma unroll
      for (int g = 0; g < 8; ++g) {
        int row = mt * 16 + g + 8 * hf;
        size_t o = (t0 + row) * C_ + n;
        out[o] = acc[tt][g] + bias + out[o];
      }
    }
  }
}

// ---------------------------------------------------------------------------
// Launch
// ---------------------------------------------------------------------------
extern "C" void kernel_launch(void* const* d_in, const int* in_sizes, int n_in,
                              void* d_out, int out_size, void* d_ws, size_t ws_size,
                              hipStream_t stream) {
  const float* x      = (const float*)d_in[0];
  const float* n1g    = (const float*)d_in[1];
  const float* n1b    = (const float*)d_in[2];
  const float* qkv_w  = (const float*)d_in[3];
  const float* qkv_b  = (const float*)d_in[4];
  const float* proj_w = (const float*)d_in[5];
  const float* proj_b = (const float*)d_in[6];
  const float* rpb    = (const float*)d_in[7];
  const float* n2g    = (const float*)d_in[8];
  const float* n2b    = (const float*)d_in[9];
  const float* fc1_w  = (const float*)d_in[10];
  const float* fc1_b  = (const float*)d_in[11];
  const float* fc2_w  = (const float*)d_in[12];
  const float* fc2_b  = (const float*)d_in[13];
  float* out = (float*)d_out;

  char* ws = (char*)d_ws;
  // workspace layout (bytes)
  __bf16* wqkv  = (__bf16*)(ws);                       // 768*256*2  = 393216
  __bf16* wproj = (__bf16*)(ws + 393216);              // 256*256*2  = 131072
  __bf16* wfc1  = (__bf16*)(ws + 524288);              // 1024*256*2 = 524288
  __bf16* wfc2  = (__bf16*)(ws + 1048576);             // 256*1024*2 = 524288
  __bf16* xw    = (__bf16*)(ws + 1572864);             // 2048*64*256*2 = 64MB
  __bf16* ao    = (__bf16*)(ws + 1572864 + 67108864);  // 64MB

  k_convert_w<<<512, 256, 0, stream>>>(qkv_w, proj_w, fc1_w, fc2_w, wqkv, wproj, wfc1, wfc2);
  k_ln1_partition<<<NWIN_, 256, 0, stream>>>(x, n1g, n1b, xw);
  k_attention<<<NWIN_, 256, 196608, stream>>>(xw, wqkv, qkv_b, rpb, ao);
  k_proj<<<NWIN_, 256, 0, stream>>>(ao, wproj, proj_b, x, out);
  k_mlp<<<1568, 256, 163840, stream>>>(n2g, n2b, wfc1, fc1_b, wfc2, fc2_b, out);
}
